// GraphAttention_75608604279195
// MI455X (gfx1250) — matrix operand order repaired and checked by reading, not compiled, not run on hardware
//
#include <hip/hip_runtime.h>
#include <hip/hip_bf16.h>

// ---------------- problem constants ----------------
#define H_     3
#define B_     16
#define N_     1024
#define F_IN   512
#define F_OUT  1024
#define NEG_SLOPE 0.2f

// ---------------- GEMM tiling ----------------
#define BM  128
#define BN  128
#define BK  32
#define LDT 40   // padded LDS row stride (bf16 elems) -> 80B rows, 16B-aligned frags, conflict-free b128

typedef __attribute__((ext_vector_type(16))) __bf16 v16bf;
typedef __attribute__((ext_vector_type(8)))  float  v8f;

union FragU { v16bf v; uint4 q[2]; };

__device__ __forceinline__ unsigned short f2bf(float f) {
  unsigned int u = __float_as_uint(f);
  u += 0x7FFFu + ((u >> 16) & 1u);          // round-to-nearest-even
  return (unsigned short)(u >> 16);
}
__device__ __forceinline__ float bf2f(unsigned short h) {
  return __uint_as_float(((unsigned int)h) << 16);
}

// ---------------- fp32 -> bf16 convert ----------------
__global__ void k_f32_to_bf16(const float* __restrict__ in,
                              unsigned short* __restrict__ out, long long n) {
  long long i = (long long)blockIdx.x * blockDim.x + threadIdx.x;
  long long stride = (long long)gridDim.x * blockDim.x;
  for (; i < n; i += stride) out[i] = f2bf(in[i]);
}

// ---------------- GEMM1: feats = x @ kernels[h]  (bf16 WMMA, double-buffered) ----
// xb: (16384, 512) bf16, kb: (H, 512, 1024) bf16, featsb: (H, 16384, 1024) bf16
__global__ __launch_bounds__(256) void k_gemm1(
    const unsigned short* __restrict__ xb,
    const unsigned short* __restrict__ kb,
    unsigned short* __restrict__ featsb) {
  __shared__ __align__(16) unsigned short As[2][BM][LDT];
  __shared__ __align__(16) unsigned short Bs[2][BN][LDT];   // B transposed: Bs[.][n][k]
  const int h    = blockIdx.z;
  const int row0 = blockIdx.y * BM;
  const int col0 = blockIdx.x * BN;
  const int t    = threadIdx.x;
  const int lane = t & 31;
  const int wave = t >> 5;
  const int wm   = wave & 3;     // 4 wave-rows of 32
  const int wn   = wave >> 2;    // 2 wave-cols of 64
  const unsigned short* Bmat = kb + (size_t)h * F_IN * F_OUT;

  // per-thread staging coordinates
  const int a_r = t >> 2;               // + {0,64}
  const int a_c = (t & 3) << 3;
  const int b_k = t >> 4;               // + {0,16}
  const int b_n = (t & 15) << 3;

  v8f acc[2][4];
#pragma unroll
  for (int i = 0; i < 2; ++i)
#pragma unroll
    for (int j = 0; j < 4; ++j)
#pragma unroll
      for (int e = 0; e < 8; ++e) acc[i][j][e] = 0.0f;

  const int NT = F_IN / BK;   // 16
  uint4 ra0, ra1, rb0, rb1;

  // prologue: tile 0 -> regs -> LDS buf 0
  ra0 = *(const uint4*)(xb + (size_t)(row0 + a_r) * F_IN + a_c);
  ra1 = *(const uint4*)(xb + (size_t)(row0 + a_r + 64) * F_IN + a_c);
  rb0 = *(const uint4*)(Bmat + (size_t)b_k * F_OUT + col0 + b_n);
  rb1 = *(const uint4*)(Bmat + (size_t)(b_k + 16) * F_OUT + col0 + b_n);
  {
    *(uint4*)&As[0][a_r][a_c] = ra0;
    *(uint4*)&As[0][a_r + 64][a_c] = ra1;
    const unsigned short* e0 = (const unsigned short*)&rb0;
    const unsigned short* e1 = (const unsigned short*)&rb1;
#pragma unroll
    for (int q = 0; q < 8; ++q) Bs[0][b_n + q][b_k] = e0[q];
#pragma unroll
    for (int q = 0; q < 8; ++q) Bs[0][b_n + q][b_k + 16] = e1[q];
  }

  for (int kt = 0; kt < NT; ++kt) {
    __syncthreads();                    // single barrier per K-tile
    const int buf = kt & 1;

    // fragments from current buffer
    v16bf af[2], bfr[4];
#pragma unroll
    for (int mi = 0; mi < 2; ++mi) {    // A: K interleave {h8..h8+7, 16+h8..}
      int ar = wm * 32 + mi * 16 + (lane & 15);
      int ah = (lane >> 4) << 3;
      FragU u;
      u.q[0] = *(const uint4*)&As[buf][ar][ah];
      u.q[1] = *(const uint4*)&As[buf][ar][16 + ah];
      af[mi] = u.v;
    }
#pragma unroll
    for (int ni = 0; ni < 4; ++ni) {    // B: lane = n, K interleave {h8..h8+7, 16+h8..}
      int bn = wn * 64 + ni * 16 + (lane & 15);
      int bk = (lane >> 4) << 3;
      FragU u;
      u.q[0] = *(const uint4*)&Bs[buf][bn][bk];
      u.q[1] = *(const uint4*)&Bs[buf][bn][bk + 16];
      bfr[ni] = u.v;
    }

    // issue next tile's global loads while WMMAs run
    const bool more = (kt + 1 < NT);
    if (more) {
      const int k1 = (kt + 1) * BK;
      ra0 = *(const uint4*)(xb + (size_t)(row0 + a_r) * F_IN + k1 + a_c);
      ra1 = *(const uint4*)(xb + (size_t)(row0 + a_r + 64) * F_IN + k1 + a_c);
      rb0 = *(const uint4*)(Bmat + (size_t)(k1 + b_k) * F_OUT + col0 + b_n);
      rb1 = *(const uint4*)(Bmat + (size_t)(k1 + b_k + 16) * F_OUT + col0 + b_n);
    }
    if (kt + 2 < NT) {                  // distance-2 prefetch toward L2 (global_prefetch_b8)
      const int k2 = (kt + 2) * BK;
      __builtin_prefetch(xb + (size_t)(row0 + a_r) * F_IN + k2 + a_c, 0, 1);
      __builtin_prefetch(Bmat + (size_t)(k2 + b_k) * F_OUT + col0 + b_n, 0, 1);
    }

#pragma unroll
    for (int mi = 0; mi < 2; ++mi)
#pragma unroll
      for (int ni = 0; ni < 4; ++ni)
        acc[mi][ni] = __builtin_amdgcn_wmma_f32_16x16x32_bf16(
            false, af[mi], false, bfr[ni], (short)0, acc[mi][ni], false, false);

    if (more) {                         // fill the other buffer (safe: one barrier back)
      const int nb = buf ^ 1;
      *(uint4*)&As[nb][a_r][a_c] = ra0;
      *(uint4*)&As[nb][a_r + 64][a_c] = ra1;
      const unsigned short* e0 = (const unsigned short*)&rb0;
      const unsigned short* e1 = (const unsigned short*)&rb1;
#pragma unroll
      for (int q = 0; q < 8; ++q) Bs[nb][b_n + q][b_k] = e0[q];
#pragma unroll
      for (int q = 0; q < 8; ++q) Bs[nb][b_n + q][b_k + 16] = e1[q];
    }
  }

  unsigned short* C = featsb + (size_t)h * (B_ * N_) * F_OUT;
  const int rb = row0 + wm * 32 + ((lane >> 4) << 3);
  const int cb = col0 + wn * 64 + (lane & 15);
#pragma unroll
  for (int mi = 0; mi < 2; ++mi)
#pragma unroll
    for (int ni = 0; ni < 4; ++ni)
#pragma unroll
      for (int r = 0; r < 8; ++r)
        C[(size_t)(rb + mi * 16 + r) * F_OUT + cb + ni * 16] = f2bf(acc[mi][ni][r]);
}

// ---------------- scores: wave-per-row dot with attn vectors ----------------
__global__ __launch_bounds__(256) void k_scores(
    const unsigned short* __restrict__ featsb,
    const float* __restrict__ attn_self, const float* __restrict__ attn_neigh,
    float* __restrict__ s_self, float* __restrict__ s_neigh) {
  int wid  = (blockIdx.x * 256 + threadIdx.x) >> 5;   // one wave32 per (h,b,n) row
  int lane = threadIdx.x & 31;
  if (wid >= H_ * B_ * N_) return;
  int h = wid / (B_ * N_);
  const unsigned short* row = featsb + (size_t)wid * F_OUT;
  const float* as = attn_self + h * F_OUT;
  const float* an = attn_neigh + h * F_OUT;
  float ss = 0.f, sn = 0.f;
#pragma unroll
  for (int i = 0; i < 4; ++i) {
    int k = i * 256 + lane * 8;
    uint4 q = *(const uint4*)(row + k);
    const unsigned short* e = (const unsigned short*)&q;
#pragma unroll
    for (int j = 0; j < 8; ++j) {
      float f = bf2f(e[j]);
      ss += f * as[k + j];
      sn += f * an[k + j];
    }
  }
#pragma unroll
  for (int off = 16; off > 0; off >>= 1) {
    ss += __shfl_down(ss, off, 32);
    sn += __shfl_down(sn, off, 32);
  }
  if (lane == 0) { s_self[wid] = ss; s_neigh[wid] = sn; }
}

// ---------------- per-(h,b) max of s_neigh (monotonic LeakyReLU => shared max) ----
__global__ void k_rowmax(const float* __restrict__ s_neigh, float* __restrict__ bmax) {
  int hb = blockIdx.x;
  const float* p = s_neigh + (size_t)hb * N_;
  float m = -3.0e38f;
  for (int j = threadIdx.x; j < N_; j += 256) m = fmaxf(m, p[j]);
  __shared__ float red[256];
  red[threadIdx.x] = m;
  __syncthreads();
  for (int s = 128; s > 0; s >>= 1) {
    if (threadIdx.x < s) red[threadIdx.x] = fmaxf(red[threadIdx.x], red[threadIdx.x + s]);
    __syncthreads();
  }
  if (threadIdx.x == 0) bmax[hb] = red[0];
}

// ---------------- softmax over rank-1 logits, writes attn fp32 into d_out --------
__global__ __launch_bounds__(256) void k_softmax(
    const float* __restrict__ s_self, const float* __restrict__ s_neigh,
    const float* __restrict__ bmax, float* __restrict__ attn) {
  const int row = blockIdx.x;        // (h*B + b)*N + i
  const int hb  = row >> 10;         // / N_
  const int t   = threadIdx.x;
  const float a = s_self[row];
  float m = a + bmax[hb];
  m = m >= 0.f ? m : NEG_SLOPE * m;  // LeakyReLU monotonic => exact row max
  const float* sn = s_neigh + (size_t)hb * N_;
  float e[4]; float sum = 0.f;
#pragma unroll
  for (int p = 0; p < 4; ++p) {
    int j = p * 256 + t;
    float v = a + sn[j];
    v = v >= 0.f ? v : NEG_SLOPE * v;
    e[p] = __expf(v - m);
    sum += e[p];
  }
  __shared__ float red[256];
  red[t] = sum;
  __syncthreads();
  for (int s = 128; s > 0; s >>= 1) {
    if (t < s) red[t] += red[t + s];
    __syncthreads();
  }
  const float inv = 1.0f / red[0];
  float* o = attn + (size_t)row * N_;
#pragma unroll
  for (int p = 0; p < 4; ++p) o[p * 256 + t] = e[p] * inv;
}

// ---------------- GEMM2: mean_h(attn[h,b] @ feats[h,b] + bias[h]) ----------------
// Double-buffered; h and k0 loops flattened into one 96-tile pipeline.
__global__ __launch_bounds__(256) void k_gemm2(
    const float* __restrict__ attn,            // (H,B,N,N) fp32 (lives in d_out)
    const unsigned short* __restrict__ featsb, // (H,B,N,F_OUT) bf16
    const float* __restrict__ biases,          // (H,F_OUT) fp32
    float* __restrict__ out2) {                // (B,N,F_OUT) fp32
  __shared__ __align__(16) unsigned short As[2][BM][LDT];
  __shared__ __align__(16) unsigned short Bs[2][BN][LDT];
  const int b    = blockIdx.z;
  const int row0 = blockIdx.y * BM;
  const int col0 = blockIdx.x * BN;
  const int t    = threadIdx.x;
  const int lane = t & 31;
  const int wave = t >> 5;
  const int wm   = wave & 3;
  const int wn   = wave >> 2;

  // per-thread staging coordinates
  const int a_r = t >> 3;               // + {0,32,64,96} (fp32 A: 4 floats per chunk)
  const int a_c = (t & 7) << 2;
  const int b_k = t >> 4;               // + {0,16}
  const int b_n = (t & 15) << 3;

  v8f acc[2][4];
#pragma unroll
  for (int i = 0; i < 2; ++i)
#pragma unroll
    for (int j = 0; j < 4; ++j)
#pragma unroll
      for (int e = 0; e < 8; ++e) acc[i][j][e] = 0.0f;

  const int NT = H_ * (N_ / BK);        // 96 tiles across all heads
  float4 fa[4];
  uint4 rb0, rb1;

  {  // prologue: tile 0 (h=0, k0=0)
    const float* Ah0 = attn + (size_t)b * N_ * N_;
    const unsigned short* Bh0 = featsb + (size_t)b * N_ * F_OUT;
#pragma unroll
    for (int p = 0; p < 4; ++p)
      fa[p] = *(const float4*)(Ah0 + (size_t)(row0 + a_r + p * 32) * N_ + a_c);
    rb0 = *(const uint4*)(Bh0 + (size_t)b_k * F_OUT + col0 + b_n);
    rb1 = *(const uint4*)(Bh0 + (size_t)(b_k + 16) * F_OUT + col0 + b_n);
#pragma unroll
    for (int p = 0; p < 4; ++p) {
      As[0][a_r + p * 32][a_c + 0] = f2bf(fa[p].x);
      As[0][a_r + p * 32][a_c + 1] = f2bf(fa[p].y);
      As[0][a_r + p * 32][a_c + 2] = f2bf(fa[p].z);
      As[0][a_r + p * 32][a_c + 3] = f2bf(fa[p].w);
    }
    const unsigned short* e0 = (const unsigned short*)&rb0;
    const unsigned short* e1 = (const unsigned short*)&rb1;
#pragma unroll
    for (int q = 0; q < 8; ++q) Bs[0][b_n + q][b_k] = e0[q];
#pragma unroll
    for (int q = 0; q < 8; ++q) Bs[0][b_n + q][b_k + 16] = e1[q];
  }

  for (int kt = 0; kt < NT; ++kt) {
    __syncthreads();
    const int buf = kt & 1;

    v16bf af[2], bfr[4];
#pragma unroll
    for (int mi = 0; mi < 2; ++mi) {
      int ar = wm * 32 + mi * 16 + (lane & 15);
      int ah = (lane >> 4) << 3;
      FragU u;
      u.q[0] = *(const uint4*)&As[buf][ar][ah];
      u.q[1] = *(const uint4*)&As[buf][ar][16 + ah];
      af[mi] = u.v;
    }
#pragma unroll
    for (int ni = 0; ni < 4; ++ni) {
      int bn = wn * 64 + ni * 16 + (lane & 15);
      int bk = (lane >> 4) << 3;
      FragU u;
      u.q[0] = *(const uint4*)&Bs[buf][bn][bk];
      u.q[1] = *(const uint4*)&Bs[buf][bn][bk + 16];
      bfr[ni] = u.v;
    }

    const bool more = (kt + 1 < NT);
    if (more) {                         // stage tile kt+1 into registers
      const int kt1 = kt + 1;
      const int h1  = kt1 >> 5;
      const int k1  = (kt1 & 31) * BK;
      const float* Ah1 = attn + ((size_t)(h1 * B_ + b)) * N_ * N_;
      const unsigned short* Bh1 = featsb + ((size_t)(h1 * B_ + b)) * N_ * F_OUT;
#pragma unroll
      for (int p = 0; p < 4; ++p)
        fa[p] = *(const float4*)(Ah1 + (size_t)(row0 + a_r + p * 32) * N_ + k1 + a_c);
      rb0 = *(const uint4*)(Bh1 + (size_t)(k1 + b_k) * F_OUT + col0 + b_n);
      rb1 = *(const uint4*)(Bh1 + (size_t)(k1 + b_k + 16) * F_OUT + col0 + b_n);
    }
    if (kt + 2 < NT) {                  // distance-2 prefetch (global_prefetch_b8)
      const int kt2 = kt + 2;
      const int h2  = kt2 >> 5;
      const int k2  = (kt2 & 31) * BK;
      const float* Ah2 = attn + ((size_t)(h2 * B_ + b)) * N_ * N_;
      const unsigned short* Bh2 = featsb + ((size_t)(h2 * B_ + b)) * N_ * F_OUT;
      __builtin_prefetch(Ah2 + (size_t)(row0 + a_r) * N_ + k2 + a_c, 0, 1);
      __builtin_prefetch(Bh2 + (size_t)(k2 + b_k) * F_OUT + col0 + b_n, 0, 1);
    }

#pragma unroll
    for (int mi = 0; mi < 2; ++mi)
#pragma unroll
      for (int ni = 0; ni < 4; ++ni)
        acc[mi][ni] = __builtin_amdgcn_wmma_f32_16x16x32_bf16(
            false, af[mi], false, bfr[ni], (short)0, acc[mi][ni], false, false);

    if (more) {
      const int nb = buf ^ 1;
#pragma unroll
      for (int p = 0; p < 4; ++p) {
        As[nb][a_r + p * 32][a_c + 0] = f2bf(fa[p].x);
        As[nb][a_r + p * 32][a_c + 1] = f2bf(fa[p].y);
        As[nb][a_r + p * 32][a_c + 2] = f2bf(fa[p].z);
        As[nb][a_r + p * 32][a_c + 3] = f2bf(fa[p].w);
      }
      const unsigned short* e0 = (const unsigned short*)&rb0;
      const unsigned short* e1 = (const unsigned short*)&rb1;
#pragma unroll
      for (int q = 0; q < 8; ++q) Bs[nb][b_n + q][b_k] = e0[q];
#pragma unroll
      for (int q = 0; q < 8; ++q) Bs[nb][b_n + q][b_k + 16] = e1[q];
    }
  }

  const float inv3 = 1.0f / 3.0f;
  const int rbase = row0 + wm * 32 + ((lane >> 4) << 3);
#pragma unroll
  for (int ni = 0; ni < 4; ++ni) {
    int col = col0 + wn * 64 + ni * 16 + (lane & 15);
    float bs = biases[col] + biases[F_OUT + col] + biases[2 * F_OUT + col];
#pragma unroll
    for (int mi = 0; mi < 2; ++mi)
#pragma unroll
      for (int r = 0; r < 8; ++r)
        out2[(size_t)(b * N_ + rbase + mi * 16 + r) * F_OUT + col] =
            (acc[mi][ni][r] + bs) * inv3;
  }
}

// ---------------- launcher ----------------
extern "C" void kernel_launch(void* const* d_in, const int* in_sizes, int n_in,
                              void* d_out, int out_size, void* d_ws, size_t ws_size,
                              hipStream_t stream) {
  (void)in_sizes; (void)n_in; (void)out_size; (void)ws_size;
  const float* x          = (const float*)d_in[0];   // (B,N,F)
  const float* kernels    = (const float*)d_in[1];   // (H,F,F_)
  const float* biases     = (const float*)d_in[2];   // (H,F_)
  const float* attn_self  = (const float*)d_in[3];   // (H,F_)
  const float* attn_neigh = (const float*)d_in[4];   // (H,F_)
  float* out = (float*)d_out;                        // (H+1,B,N,N)

  const long long nX = (long long)B_ * N_ * F_IN;          //  8,388,608
  const long long nK = (long long)H_ * F_IN * F_OUT;       //  1,572,864
  const long long nFeat = (long long)H_ * B_ * N_ * F_OUT; // 50,331,648
  const long long nAttn = (long long)H_ * B_ * N_ * N_;    // 50,331,648

  char* ws = (char*)d_ws;
  unsigned short* xb     = (unsigned short*)ws;                       // 16 MB
  unsigned short* kb     = (unsigned short*)(ws + 2 * nX);            //  3 MB
  unsigned short* featsb = (unsigned short*)(ws + 2 * (nX + nK));     // 100 MB
  float* s_self  = (float*)(ws + 2 * (nX + nK + nFeat));
  float* s_neigh = s_self + H_ * B_ * N_;
  float* bmax    = s_neigh + H_ * B_ * N_;

  k_f32_to_bf16<<<dim3(8192), 256, 0, stream>>>(x, xb, nX);
  k_f32_to_bf16<<<dim3(1536), 256, 0, stream>>>(kernels, kb, nK);

  k_gemm1<<<dim3(F_OUT / BN, (B_ * N_) / BM, H_), 256, 0, stream>>>(xb, kb, featsb);

  k_scores<<<dim3((H_ * B_ * N_) / 8), 256, 0, stream>>>(featsb, attn_self, attn_neigh,
                                                         s_self, s_neigh);
  k_rowmax<<<dim3(H_ * B_), 256, 0, stream>>>(s_neigh, bmax);
  k_softmax<<<dim3(H_ * B_ * N_), 256, 0, stream>>>(s_self, s_neigh, bmax, out);

  k_gemm2<<<dim3(F_OUT / BN, N_ / BM, B_), 256, 0, stream>>>(out, featsb, biases,
                                                             out + nAttn);
}
